// CausalSelfAttention_65103114273084
// MI455X (gfx1250) — matrix operand order repaired and checked by reading, not compiled, not run on hardware
//
#include <hip/hip_runtime.h>
#include <hip/hip_bf16.h>

// ---------------------------------------------------------------------------
// CausalSelfAttention forward for MI455X (gfx1250, wave32, bf16 WMMA)
// B=8, T=1024, C=2048, NH=16, D=128
// ---------------------------------------------------------------------------

typedef __attribute__((ext_vector_type(16))) __bf16 v16bf;
typedef __attribute__((ext_vector_type(8)))  __bf16 v8bf;
typedef __attribute__((ext_vector_type(8)))  float  v8f;
typedef int v4i_vs __attribute__((vector_size(16)));   // matches builtin param type

#define Bc   8
#define Tc   1024
#define Cc   2048
#define NHc  16
#define Dc   128
#define N3c  6144
#define Mc   8192   // B*T

// ---- CDNA5 async global->LDS copy (ASYNCcnt path), guarded ----------------
#if __has_builtin(__builtin_amdgcn_global_load_async_to_lds_b128)
#define USE_ASYNC_LDS 1
#else
#define USE_ASYNC_LDS 0
#endif

__device__ __forceinline__ void async_copy_b128(const void* gsrc, void* ldst) {
#if USE_ASYNC_LDS
    __builtin_amdgcn_global_load_async_to_lds_b128(
        (v4i_vs*)gsrc, (v4i_vs*)ldst, 0, 0);
#else
    *(uint4*)ldst = *(const uint4*)gsrc;
#endif
}

__device__ __forceinline__ void async_publish_wait() {
#if USE_ASYNC_LDS
#if __has_builtin(__builtin_amdgcn_s_wait_asynccnt)
    __builtin_amdgcn_s_wait_asynccnt(0);
#else
    asm volatile("s_wait_asynccnt 0x0" ::: "memory");
#endif
#endif
}

__device__ __forceinline__ v8f wmma_bf16(v16bf a, v16bf b, v8f c) {
    return __builtin_amdgcn_wmma_f32_16x16x32_bf16(
        /*neg_a=*/false, a, /*neg_b=*/false, b,
        /*c_mod=*/(short)0, c, /*reuse_a=*/false, /*reuse_b=*/false);
}

// A-matrix fragment (16x32 bf16). Per CDNA5 ISA: lane l (M=l), lane-half hi:
// packed K order = [hi*8 .. hi*8+7] then [16+hi*8 .. 16+hi*8+7] -> two 16B runs.
__device__ __forceinline__ v16bf ld_fragA(const __bf16* row, int hi) {
    const v8bf* p = (const v8bf*)(row + hi * 8);
    v8bf x = p[0];           // elements [hi*8, hi*8+8)
    v8bf y = p[2];           // elements [16+hi*8, 16+hi*8+8)
    v16bf r;
#pragma unroll
    for (int i = 0; i < 8; ++i) { r[i] = x[i]; r[i + 8] = y[i]; }
    return r;
}

// B-matrix fragment (32x16 bf16), row = the N-index row of a [n][k] LDS tile.
// Lane l (N=l), lane-half hi holds K = [hi*8 .. hi*8+7] then [16+hi*8 .. 16+hi*8+7].
__device__ __forceinline__ v16bf ld_fragB(const __bf16* row, int hi) {
    const v8bf* p = (const v8bf*)(row + hi * 8);
    v8bf x = p[0];
    v8bf y = p[2];
    v16bf r;
#pragma unroll
    for (int i = 0; i < 8; ++i) { r[i] = x[i]; r[i + 8] = y[i]; }
    return r;
}

// ---------------------------------------------------------------------------
// Kernel 1: qkv = x @ W_attn + b_attn, scattered to Q/K/V [B,NH,T,D] bf16.
// Block tile 128x128, K-step 32, 8 waves (2x4 wave grid), double-buffered LDS.
// ---------------------------------------------------------------------------
__global__ __launch_bounds__(256) void qkv_gemm_kernel(
    const float* __restrict__ X, const float* __restrict__ W,
    const float* __restrict__ bias,
    __bf16* __restrict__ Qd, __bf16* __restrict__ Kd, __bf16* __restrict__ Vd)
{
    __shared__ __align__(16) __bf16 As[2][128][32];   // [m][k]
    __shared__ __align__(16) __bf16 Bs[2][128][32];   // [n][k] (transposed)

    const int tid  = threadIdx.x;
    const int wave = tid >> 5, lane = tid & 31;
    const int l = lane & 15, hi = lane >> 4;
    const int wm = wave & 1, wn = wave >> 1;            // 2 x 4 wave grid
    const int m0 = blockIdx.x * 128, n0 = blockIdx.y * 128;

    v8f acc[4][2] = {};

    auto stage = [&](int k0, int buf) {
#pragma unroll
        for (int i = 0; i < 4; ++i) {
            int lin = tid + i * 256;                    // [0,1024)
            int r = lin >> 3, c = (lin & 7) * 4;
            const float4 v = *(const float4*)&X[(size_t)(m0 + r) * Cc + k0 + c];
            As[buf][r][c + 0] = (__bf16)v.x; As[buf][r][c + 1] = (__bf16)v.y;
            As[buf][r][c + 2] = (__bf16)v.z; As[buf][r][c + 3] = (__bf16)v.w;
        }
#pragma unroll
        for (int i = 0; i < 4; ++i) {
            int lin = tid + i * 256;                    // [0,1024)
            int k = lin >> 5, n4 = (lin & 31) * 4;
            const float4 v = *(const float4*)&W[(size_t)(k0 + k) * N3c + n0 + n4];
            Bs[buf][n4 + 0][k] = (__bf16)v.x; Bs[buf][n4 + 1][k] = (__bf16)v.y;
            Bs[buf][n4 + 2][k] = (__bf16)v.z; Bs[buf][n4 + 3][k] = (__bf16)v.w;
        }
    };

    const int NK = Cc / 32;
    stage(0, 0);
    for (int kt = 0; kt < NK; ++kt) {
        __syncthreads();                       // publish buf kt&1 / retire kt-1
        if (kt + 1 < NK) {
            stage((kt + 1) * 32, (kt + 1) & 1);
            if (kt + 2 < NK) {                 // prefetch weight stream +2 steps
                __builtin_prefetch(
                    &W[(size_t)((kt + 2) * 32 + (tid >> 3)) * N3c + n0 + (tid & 7) * 16],
                    0, 1);
            }
        }
        const int buf = kt & 1;
        v16bf af[4], bfr[2];
#pragma unroll
        for (int mf = 0; mf < 4; ++mf)
            af[mf] = ld_fragA(&As[buf][wm * 64 + mf * 16 + l][0], hi);
#pragma unroll
        for (int nf = 0; nf < 2; ++nf)
            bfr[nf] = ld_fragB(&Bs[buf][wn * 32 + nf * 16 + l][0], hi);
#pragma unroll
        for (int mf = 0; mf < 4; ++mf)
#pragma unroll
            for (int nf = 0; nf < 2; ++nf)
                acc[mf][nf] = wmma_bf16(af[mf], bfr[nf], acc[mf][nf]);
    }

    // epilogue: add bias, scatter into per-head Q/K/V layouts
#pragma unroll
    for (int mf = 0; mf < 4; ++mf) {
#pragma unroll
        for (int nf = 0; nf < 2; ++nf) {
#pragma unroll
            for (int r = 0; r < 8; ++r) {
                int grow = m0 + wm * 64 + mf * 16 + r + hi * 8;   // [0,8192)
                int gcol = n0 + wn * 32 + nf * 16 + l;            // [0,6144)
                float val = acc[mf][nf][r] + bias[gcol];
                int which = gcol >> 11;            // 0=q 1=k 2=v
                int c = gcol & (Cc - 1);
                int h = c >> 7, d = c & (Dc - 1);
                int b = grow >> 10, t = grow & (Tc - 1);
                __bf16* dst = (which == 0) ? Qd : ((which == 1) ? Kd : Vd);
                dst[((((size_t)b * NHc + h) * Tc + t) << 7) + d] = (__bf16)val;
            }
        }
    }
}

// ---------------------------------------------------------------------------
// Kernel 2: flash attention. Grid (T/64, NH, B), 4 waves; wave w owns 16
// queries. Online softmax, key tiles of 32. Q/K staged via async-to-LDS.
// ---------------------------------------------------------------------------
__global__ __launch_bounds__(128) void attn_kernel(
    const __bf16* __restrict__ Qg, const __bf16* __restrict__ Kg,
    const __bf16* __restrict__ Vg, __bf16* __restrict__ Y)
{
    __shared__ __align__(16) __bf16 Qs[64][128];
    __shared__ __align__(16) __bf16 Ks[32][128];
    __shared__ __align__(16) __bf16 VTs[128][32];   // [d][key]
    __shared__ __align__(16) __bf16 Ps[4][16][32];  // per-wave P staging

    const int tid  = threadIdx.x;
    const int wave = tid >> 5, lane = tid & 31;
    const int l = lane & 15, hi = lane >> 4;
    const int qt = blockIdx.x, h = blockIdx.y, b = blockIdx.z;
    const size_t bh = (size_t)b * NHc + h;
    const __bf16* Qh = Qg + bh * Tc * Dc;
    const __bf16* Kh = Kg + bh * Tc * Dc;
    const __bf16* Vh = Vg + bh * Tc * Dc;
    const int q0blk = qt * 64;
    const int q0 = q0blk + wave * 16;

    // stage all 64 queries once (async -> LDS)
#pragma unroll
    for (int i = 0; i < 8; ++i) {
        int lin = tid + i * 128;                     // [0,1024)
        int r = lin >> 4, c8 = (lin & 15) * 8;
        async_copy_b128(&Qh[(size_t)(q0blk + r) * Dc + c8], &Qs[r][c8]);
    }

    v8f o[8] = {};
    float row_m[8], row_l[8];
#pragma unroll
    for (int r = 0; r < 8; ++r) { row_m[r] = -1.0e30f; row_l[r] = 0.0f; }

    const float sc = 0.08838834764831845f;           // 1/sqrt(128)
    const int nkv = (q0blk + 64) / 32;               // causal limit of block

    for (int kvt = 0; kvt < nkv; ++kvt) {
        const int kv0 = kvt * 32;
        __syncthreads();                              // retire prior LDS reads
        // stage K tile row-major (async -> LDS)
#pragma unroll
        for (int i = 0; i < 4; ++i) {
            int lin = tid + i * 128;                 // [0,512)
            int r = lin >> 4, c8 = (lin & 15) * 8;
            async_copy_b128(&Kh[(size_t)(kv0 + r) * Dc + c8], &Ks[r][c8]);
        }
        // stage V tile transposed -> VTs[d][key] (through VGPRs)
#pragma unroll
        for (int i = 0; i < 4; ++i) {
            int lin = tid + i * 128;
            int key = lin >> 4, d8 = (lin & 15) * 8;
            v8bf v = *(const v8bf*)&Vh[(size_t)(kv0 + key) * Dc + d8];
#pragma unroll
            for (int j = 0; j < 8; ++j) VTs[d8 + j][key] = v[j];
        }
        async_publish_wait();                         // ASYNCcnt -> 0
        __syncthreads();

        if (kv0 <= q0 + 15) {                        // wave-uniform: any key valid
            // S = Q K^T for 16 queries x 32 keys (two 16x16 accum halves)
            v8f s0 = {}, s1 = {};
#pragma unroll
            for (int kc = 0; kc < 4; ++kc) {
                v16bf aq  = ld_fragA(&Qs[wave * 16 + l][kc * 32], hi);
                v16bf bk0 = ld_fragB(&Ks[l][kc * 32], hi);
                v16bf bk1 = ld_fragB(&Ks[16 + l][kc * 32], hi);
                s0 = wmma_bf16(aq, bk0, s0);
                s1 = wmma_bf16(aq, bk1, s1);
            }
            // scale + causal mask + online softmax (row m = r + 8*hi lives in
            // lanes of one 16-lane half -> shfl_xor masks 1,2,4,8 stay in-half)
            float alpha[8];
#pragma unroll
            for (int r = 0; r < 8; ++r) {
                int qrow = q0 + r + hi * 8;
                float v0 = s0[r] * sc; if (kv0 + l      > qrow) v0 = -1.0e30f;
                float v1 = s1[r] * sc; if (kv0 + 16 + l > qrow) v1 = -1.0e30f;
                float mx = fmaxf(v0, v1);
                mx = fmaxf(mx, __shfl_xor(mx, 1));
                mx = fmaxf(mx, __shfl_xor(mx, 2));
                mx = fmaxf(mx, __shfl_xor(mx, 4));
                mx = fmaxf(mx, __shfl_xor(mx, 8));
                float nm = fmaxf(row_m[r], mx);
                float al = __expf(row_m[r] - nm);
                float p0 = __expf(v0 - nm);
                float p1 = __expf(v1 - nm);
                float ps = p0 + p1;
                ps += __shfl_xor(ps, 1);
                ps += __shfl_xor(ps, 2);
                ps += __shfl_xor(ps, 4);
                ps += __shfl_xor(ps, 8);
                row_l[r] = row_l[r] * al + ps;
                row_m[r] = nm;
                alpha[r] = al;
                // stage P in A-matrix tile layout [m][key]
                Ps[wave][r + hi * 8][l]      = (__bf16)p0;
                Ps[wave][r + hi * 8][16 + l] = (__bf16)p1;
            }
            // rescale running output
#pragma unroll
            for (int df = 0; df < 8; ++df)
#pragma unroll
                for (int r = 0; r < 8; ++r) o[df][r] *= alpha[r];

            // same-wave DS RAW: wait + compiler memory barrier
            asm volatile("s_wait_dscnt 0" ::: "memory");

            v16bf ap = ld_fragA(&Ps[wave][l][0], hi);
#pragma unroll
            for (int df = 0; df < 8; ++df) {
                v16bf bv = ld_fragB(&VTs[df * 16 + l][0], hi);
                o[df] = wmma_bf16(ap, bv, o[df]);
            }
        }
    }

    // finalize: divide by row sum, write y in [B,T,C] bf16 (c = h*128 + d)
#pragma unroll
    for (int r = 0; r < 8; ++r) {
        float inv = 1.0f / row_l[r];
        int qg = q0 + r + hi * 8;
#pragma unroll
        for (int df = 0; df < 8; ++df) {
            int c = h * Dc + df * 16 + l;
            Y[((size_t)b * Tc + qg) * Cc + c] = (__bf16)(o[df][r] * inv);
        }
    }
}

// ---------------------------------------------------------------------------
// Kernel 3: out = y_bf16 @ W_proj + b_proj (fp32 out). Double-buffered;
// A tile staged with async global->LDS b128 copies.
// ---------------------------------------------------------------------------
__global__ __launch_bounds__(256) void proj_gemm_kernel(
    const __bf16* __restrict__ Yb, const float* __restrict__ W,
    const float* __restrict__ bias, float* __restrict__ Out)
{
    __shared__ __align__(16) __bf16 As[2][128][32];
    __shared__ __align__(16) __bf16 Bs[2][128][32];

    const int tid  = threadIdx.x;
    const int wave = tid >> 5, lane = tid & 31;
    const int l = lane & 15, hi = lane >> 4;
    const int wm = wave & 1, wn = wave >> 1;
    const int m0 = blockIdx.x * 128, n0 = blockIdx.y * 128;

    v8f acc[4][2] = {};

    auto stage = [&](int k0, int buf) {
        // A tile: bf16 async copies (16B chunks)
#pragma unroll
        for (int i = 0; i < 2; ++i) {
            int lin = tid + i * 256;                 // [0,512)
            int r = lin >> 2, c8 = (lin & 3) * 8;
            async_copy_b128(&Yb[(size_t)(m0 + r) * Cc + k0 + c8], &As[buf][r][c8]);
        }
        // B tile transposed (fp32 -> bf16)
#pragma unroll
        for (int i = 0; i < 4; ++i) {
            int lin = tid + i * 256;
            int k = lin >> 5, n4 = (lin & 31) * 4;
            const float4 v = *(const float4*)&W[(size_t)(k0 + k) * Cc + n0 + n4];
            Bs[buf][n4 + 0][k] = (__bf16)v.x; Bs[buf][n4 + 1][k] = (__bf16)v.y;
            Bs[buf][n4 + 2][k] = (__bf16)v.z; Bs[buf][n4 + 3][k] = (__bf16)v.w;
        }
    };

    const int NK = Cc / 32;
    stage(0, 0);
    for (int kt = 0; kt < NK; ++kt) {
        async_publish_wait();                  // ASYNCcnt -> 0 for staged A
        __syncthreads();
        if (kt + 1 < NK) {
            stage((kt + 1) * 32, (kt + 1) & 1);
            if (kt + 2 < NK) {
                __builtin_prefetch(
                    &W[(size_t)((kt + 2) * 32 + (tid >> 3)) * Cc + n0 + (tid & 7) * 16],
                    0, 1);
            }
        }
        const int buf = kt & 1;
        v16bf af[4], bfr[2];
#pragma unroll
        for (int mf = 0; mf < 4; ++mf)
            af[mf] = ld_fragA(&As[buf][wm * 64 + mf * 16 + l][0], hi);
#pragma unroll
        for (int nf = 0; nf < 2; ++nf)
            bfr[nf] = ld_fragB(&Bs[buf][wn * 32 + nf * 16 + l][0], hi);
#pragma unroll
        for (int mf = 0; mf < 4; ++mf)
#pragma unroll
            for (int nf = 0; nf < 2; ++nf)
                acc[mf][nf] = wmma_bf16(af[mf], bfr[nf], acc[mf][nf]);
    }

#pragma unroll
    for (int mf = 0; mf < 4; ++mf)
#pragma unroll
        for (int nf = 0; nf < 2; ++nf)
#pragma unroll
            for (int r = 0; r < 8; ++r) {
                int grow = m0 + wm * 64 + mf * 16 + r + hi * 8;
                int gcol = n0 + wn * 32 + nf * 16 + l;
                Out[(size_t)grow * Cc + gcol] = acc[mf][nf][r] + bias[gcol];
            }
}

// ---------------------------------------------------------------------------
extern "C" void kernel_launch(void* const* d_in, const int* in_sizes, int n_in,
                              void* d_out, int out_size, void* d_ws, size_t ws_size,
                              hipStream_t stream) {
    (void)in_sizes; (void)n_in; (void)out_size; (void)ws_size;
    const float* x      = (const float*)d_in[0];
    const float* W_attn = (const float*)d_in[1];
    const float* b_attn = (const float*)d_in[2];
    const float* W_proj = (const float*)d_in[3];
    const float* b_proj = (const float*)d_in[4];
    float* out = (float*)d_out;

    const size_t HSZ = (size_t)Bc * NHc * Tc * Dc;   // 16.78M elems
    __bf16* Q  = (__bf16*)d_ws;
    __bf16* K  = Q + HSZ;
    __bf16* V  = K + HSZ;
    __bf16* Yb = V + HSZ;                            // total ~134 MB bf16

    qkv_gemm_kernel<<<dim3(Mc / 128, N3c / 128), dim3(256), 0, stream>>>(
        x, W_attn, b_attn, Q, K, V);
    attn_kernel<<<dim3(Tc / 64, NHc, Bc), dim3(128), 0, stream>>>(Q, K, V, Yb);
    proj_gemm_kernel<<<dim3(Mc / 128, Cc / 128), dim3(256), 0, stream>>>(
        Yb, W_proj, b_proj, out);
}
